// GCHAAttention_32839319945801
// MI455X (gfx1250) — matrix operand order repaired and checked by reading, not compiled, not run on hardware
//
#include <hip/hip_runtime.h>
#include <hip/hip_bf16.h>

typedef __attribute__((ext_vector_type(16))) _Float16 v16h;
typedef __attribute__((ext_vector_type(8)))  _Float16 v8h;
typedef __attribute__((ext_vector_type(8)))  float    v8f;

#define B_     8
#define C_     512
#define N_     1024
#define HEADS_ 8
#define D_     64
#define NEGV  (-1.0e9f)

__device__ __forceinline__ v16h mk16(v8h lo, v8h hi){
  v16h r;
#pragma unroll
  for (int i=0;i<8;++i){ r[i]=lo[i]; r[i+8]=hi[i]; }
  return r;
}
__device__ __forceinline__ float sigm(float v){ return 1.0f/(1.0f+__expf(-v)); }

// ---------------- fp32 -> fp16 weight convert ----------------
__global__ __launch_bounds__(256) void k_w2h(const float* __restrict__ src,
                                             _Float16* __restrict__ dst, int n){
  int i = blockIdx.x*256 + threadIdx.x;
  if (i < n) dst[i] = (_Float16)src[i];
}

// ---------------- pooled[b,c] = mean_n x[b,c,n] ----------------
__global__ __launch_bounds__(256) void k_pool(const float* __restrict__ x,
                                              float* __restrict__ pooled){
  __shared__ float red[256];
  int bc = blockIdx.x;
  const float* xp = x + (size_t)bc * N_;
  int t = threadIdx.x;
  float s = 0.f;
#pragma unroll
  for (int i=0;i<4;++i) s += xp[t + i*256];
  red[t] = s; __syncthreads();
  for (int st=128; st>0; st>>=1){ if (t<st) red[t]+=red[t+st]; __syncthreads(); }
  if (t==0) pooled[bc] = red[0] * (1.0f/(float)N_);
}

// ---------------- SE block: cw = sigmoid(relu(p@W1^T+b1)@W2^T+b2) ----------------
__global__ __launch_bounds__(256) void k_se(const float* __restrict__ pooled,
                                            const float* __restrict__ Wse1, const float* __restrict__ bse1,
                                            const float* __restrict__ Wse2, const float* __restrict__ bse2,
                                            float* __restrict__ cw){
  __shared__ float pl[C_];
  __shared__ float h1[32];
  int b = blockIdx.x, t = threadIdx.x;
  pl[t]     = pooled[b*C_ + t];
  pl[t+256] = pooled[b*C_ + t + 256];
  __syncthreads();
  if (t < 32){
    float s = bse1[t];
    for (int c=0;c<C_;++c) s += pl[c]*Wse1[t*C_ + c];
    h1[t] = fmaxf(s, 0.f);
  }
  __syncthreads();
  for (int o=t;o<C_;o+=256){
    float s = bse2[o];
#pragma unroll
    for (int j=0;j<32;++j) s += h1[j]*Wse2[o*32 + j];
    cw[b*C_ + o] = sigm(s);
  }
}

// ---------------- sp projection + sigmoid -> mbuf[b,h,n] ----------------
__global__ __launch_bounds__(256) void k_sp(const float* __restrict__ x,
                                            const float* __restrict__ Wsp,
                                            const float* __restrict__ bsp,
                                            float* __restrict__ mbuf){
  int p = blockIdx.x*256 + threadIdx.x;
  int b = p >> 10, n = p & (N_-1);
  const float* xp = x + ((size_t)b*C_)*N_ + n;
  float acc[HEADS_];
#pragma unroll
  for (int hh=0;hh<HEADS_;++hh) acc[hh] = 0.f;
  for (int c=0;c<C_;++c){
    float xv = xp[(size_t)c * N_];
#pragma unroll
    for (int hh=0;hh<HEADS_;++hh) acc[hh] += xv * Wsp[hh*C_ + c];
  }
#pragma unroll
  for (int hh=0;hh<HEADS_;++hh)
    mbuf[((size_t)b*HEADS_ + hh)*N_ + n] = sigm(acc[hh] + bsp[hh]);
}

// ---------------- adaptive 7x7 pool -> threshold -> expand to maskf[b,h,n] ----------------
__global__ __launch_bounds__(256) void k_mask(const float* __restrict__ mbuf,
                                              float* __restrict__ maskf){
  __shared__ float mg[7][7];
  int h = blockIdx.x, b = blockIdx.y, t = threadIdx.x;
  const float* mp = mbuf + ((size_t)b*HEADS_ + h)*N_;
  if (t < 49){
    int gi = t / 7, gj = t % 7;
    int r0 = (gi*32)/7, r1 = ((gi+1)*32 + 6)/7;
    int c0 = (gj*32)/7, c1 = ((gj+1)*32 + 6)/7;
    float s = 0.f;
    for (int r=r0;r<r1;++r)
      for (int c=c0;c<c1;++c) s += mp[r*32 + c];
    s /= (float)((r1-r0)*(c1-c0));
    mg[gi][gj] = (s > 0.5f) ? 1.0f : 0.0f;
  }
  __syncthreads();
  for (int n=t;n<N_;n+=256){
    int y = n >> 5, xw = n & 31;
    maskf[((size_t)b*HEADS_ + h)*N_ + n] = mg[(y*7)>>5][(xw*7)>>5];
  }
}

// ---------------- xf = transpose(x * cw): (B,C,N) -> (B,N,C), f32 + f16 copies ----------------
__global__ void k_xf(const float* __restrict__ x, const float* __restrict__ cw,
                     float* __restrict__ xf32, _Float16* __restrict__ xf16){
  __shared__ float tile[32][33];
  int b = blockIdx.z, c0 = blockIdx.x*32, n0 = blockIdx.y*32;
  int tx = threadIdx.x, ty = threadIdx.y;
#pragma unroll
  for (int k=0;k<4;++k){
    int c = c0 + ty + k*8;
    tile[ty + k*8][tx] = x[((size_t)b*C_ + c)*N_ + n0 + tx] * cw[b*C_ + c];
  }
  __syncthreads();
#pragma unroll
  for (int k=0;k<4;++k){
    int n = n0 + ty + k*8;
    float v = tile[tx][ty + k*8];
    size_t idx = ((size_t)b*N_ + n)*C_ + c0 + tx;
    xf32[idx] = v;
    xf16[idx] = (_Float16)v;
  }
}

// ---------------- WMMA GEMM (double-buffered LDS): out[m,o] = sum_k A[m,k]*Wt[o,k] ----------------
// mode 0: out16 = (f16)(acc + bias)   mode 1: outf = acc + bias + resid
__global__ __launch_bounds__(256) void k_gemm(const _Float16* __restrict__ A,
                                              const _Float16* __restrict__ Wt,
                                              const float* __restrict__ bias,
                                              _Float16* __restrict__ out16,
                                              float* __restrict__ outf,
                                              const float* __restrict__ resid,
                                              int mode){
  __shared__ _Float16 As[2][128][40];   // 32 K cols, padded for align + banks
  __shared__ _Float16 Bs[2][128][40];
  const int tid  = threadIdx.x;
  const int lane = tid & 31, wave = tid >> 5;
  const int m0 = blockIdx.y * 128, o0 = blockIdx.x * 128;
  const int wm = (wave >> 2) * 64, wo = (wave & 3) * 32;
  const v8f vzero = {};
  v8f acc[4][2];
#pragma unroll
  for (int i=0;i<4;++i)
#pragma unroll
    for (int j=0;j<2;++j) acc[i][j] = vzero;

  const int lr = tid >> 1, ls = (tid & 1) * 16;
  const int arow_l = lane & 15;
  const int ac0 = (lane < 16) ? 0 : 8;
  const int bc0 = (lane < 16) ? 0 : 8;
  const _Float16* aptr = A  + ((size_t)(m0+lr))*C_ + ls;
  const _Float16* bptr = Wt + ((size_t)(o0+lr))*C_ + ls;

  // preload tile 0
  {
    const uint4* a4 = (const uint4*)aptr;
    const uint4* b4 = (const uint4*)bptr;
    uint4 av0=a4[0], av1=a4[1], bv0=b4[0], bv1=b4[1];
    *(uint4*)&As[0][lr][ls]   = av0; *(uint4*)&As[0][lr][ls+8] = av1;
    *(uint4*)&Bs[0][lr][ls]   = bv0; *(uint4*)&Bs[0][lr][ls+8] = bv1;
  }
  __syncthreads();

  const int NT = C_ / 32;
  for (int it = 0; it < NT; ++it){
    const int cur = it & 1, nxt = cur ^ 1;
    const bool more = (it + 1) < NT;
    uint4 av0, av1, bv0, bv1;
    if (more){
      const int k0 = (it + 1) * 32;
      const uint4* a4 = (const uint4*)(aptr + k0);
      const uint4* b4 = (const uint4*)(bptr + k0);
      av0=a4[0]; av1=a4[1]; bv0=b4[0]; bv1=b4[1];
      if (it + 2 < NT){
        __builtin_prefetch(aptr + k0 + 32, 0, 3);   // WGP-scope prefetch of tile it+2
        __builtin_prefetch(bptr + k0 + 32, 0, 3);
      }
    }

    // compute on current LDS buffers while next-tile global loads are in flight
    v16h af[4], bf[2];
#pragma unroll
    for (int mt=0;mt<4;++mt){
      int r = wm + mt*16 + arow_l;
      v8h lo = *(const v8h*)&As[cur][r][ac0];
      v8h hi = *(const v8h*)&As[cur][r][ac0+16];
      af[mt] = mk16(lo,hi);
    }
#pragma unroll
    for (int nt=0;nt<2;++nt){
      int r = wo + nt*16 + arow_l;
      v8h lo = *(const v8h*)&Bs[cur][r][bc0];
      v8h hi = *(const v8h*)&Bs[cur][r][bc0+16];
      bf[nt] = mk16(lo,hi);
    }
#pragma unroll
    for (int mt=0;mt<4;++mt)
#pragma unroll
      for (int nt=0;nt<2;++nt)
        acc[mt][nt] = __builtin_amdgcn_wmma_f32_16x16x32_f16(
            false, af[mt], false, bf[nt], (short)0, acc[mt][nt], false, false);

    if (more){
      *(uint4*)&As[nxt][lr][ls]   = av0; *(uint4*)&As[nxt][lr][ls+8] = av1;
      *(uint4*)&Bs[nxt][lr][ls]   = bv0; *(uint4*)&Bs[nxt][lr][ls+8] = bv1;
      __syncthreads();
    }
  }

  const int ncol  = lane & 15;
  const int mhalf = (lane < 16) ? 0 : 8;
#pragma unroll
  for (int mt=0;mt<4;++mt){
#pragma unroll
    for (int nt=0;nt<2;++nt){
      int Ob = o0 + wo + nt*16 + ncol;
      float bval = bias[Ob];
#pragma unroll
      for (int r=0;r<8;++r){
        size_t idx = ((size_t)(m0 + wm + mt*16 + mhalf + r))*C_ + Ob;
        float v = acc[mt][nt][r] + bval;
        if (mode == 0) out16[idx] = (_Float16)v;
        else           outf[idx]  = v + resid[idx];
      }
    }
  }
}

// ---------------- Flash attention per (b, h, 128-query tile) ----------------
__global__ __launch_bounds__(256) void k_attn(const _Float16* __restrict__ Qh,
                                              const _Float16* __restrict__ Kh,
                                              const _Float16* __restrict__ Vh,
                                              const float* __restrict__ maskf,
                                              _Float16* __restrict__ attn16){
  __shared__ _Float16 Ks[128][72];     // [key][d], padded
  __shared__ _Float16 Vt[64][136];     // [d][key], padded
  __shared__ _Float16 Ps[8][16][40];   // per-wave P staging (16 rows x 32 keys)
  const int tid = threadIdx.x, lane = tid & 31, wave = tid >> 5;
  const int qt = blockIdx.x, h = blockIdx.y, b = blockIdx.z;
  const int q0 = qt*128 + wave*16;
  const size_t bh = (size_t)b*HEADS_ + h;
  const int l15   = lane & 15;
  const int half8 = (lane < 16) ? 0 : 8;
  const int ac0   = (lane < 16) ? 0 : 8;
  const int bc0   = (lane < 16) ? 0 : 8;
  const v8f vzero = {};

  // Q fragments (d=64 -> 2 K-steps of 32), pinned in registers
  v16h qa[2];
  {
    const _Float16* qrow = Qh + ((size_t)b*N_ + q0 + l15)*C_ + h*D_;
#pragma unroll
    for (int kk=0;kk<2;++kk){
      v8h lo = *(const v8h*)(qrow + kk*32 + ac0);
      v8h hi = *(const v8h*)(qrow + kk*32 + ac0 + 16);
      qa[kk] = mk16(lo,hi);
    }
  }
  float mq[8];
#pragma unroll
  for (int r=0;r<8;++r) mq[r] = maskf[bh*N_ + q0 + half8 + r];

  float mrow[8], lrow[8];
  v8f oacc[4];
#pragma unroll
  for (int r=0;r<8;++r){ mrow[r] = -1.0e30f; lrow[r] = 0.f; }
#pragma unroll
  for (int d=0;d<4;++d) oacc[d] = vzero;

  for (int kb=0; kb<N_; kb+=128){
    __syncthreads();
    { // stage K (row-major) and V (transposed) tiles
      const int key = tid >> 1, s0 = (tid & 1)*32;
      const uint4* k4 = (const uint4*)(Kh + ((size_t)b*N_ + kb + key)*C_ + h*D_ + s0);
      uint4 kv0=k4[0], kv1=k4[1], kv2=k4[2], kv3=k4[3];
      *(uint4*)&Ks[key][s0]    = kv0;
      *(uint4*)&Ks[key][s0+8]  = kv1;
      *(uint4*)&Ks[key][s0+16] = kv2;
      *(uint4*)&Ks[key][s0+24] = kv3;
      union { uint4 u[4]; _Float16 hval[32]; } vu;
      const uint4* v4 = (const uint4*)(Vh + ((size_t)b*N_ + kb + key)*C_ + h*D_ + s0);
      vu.u[0]=v4[0]; vu.u[1]=v4[1]; vu.u[2]=v4[2]; vu.u[3]=v4[3];
#pragma unroll
      for (int i=0;i<32;++i) Vt[s0+i][key] = vu.hval[i];
      if (kb + 128 < N_){
        __builtin_prefetch(Kh + ((size_t)b*N_ + kb + 128 + key)*C_ + h*D_ + s0, 0, 3);
        __builtin_prefetch(Vh + ((size_t)b*N_ + kb + 128 + key)*C_ + h*D_ + s0, 0, 3);
      }
    }
    __syncthreads();

    // S = Q K^T for 8 key subtiles
    v8f sacc[8];
#pragma unroll
    for (int nt=0;nt<8;++nt){
      const int keyl = nt*16 + l15;
      v8f c = vzero;
#pragma unroll
      for (int kk=0;kk<2;++kk){
        v8h lo = *(const v8h*)&Ks[keyl][kk*32 + bc0];
        v8h hi = *(const v8h*)&Ks[keyl][kk*32 + bc0 + 16];
        v16h bfr = mk16(lo,hi);
        c = __builtin_amdgcn_wmma_f32_16x16x32_f16(false, qa[kk], false, bfr, (short)0, c, false, false);
      }
      sacc[nt] = c;
    }

    // scale + mask bias + online softmax (half-wave row reductions)
    float newm[8];
#pragma unroll
    for (int r=0;r<8;++r) newm[r] = mrow[r];
#pragma unroll
    for (int nt=0;nt<8;++nt){
      float mk = maskf[bh*N_ + kb + nt*16 + l15];
#pragma unroll
      for (int r=0;r<8;++r){
        float v = sacc[nt][r]*0.125f + (1.0f - mq[r]*mk)*NEGV;
        sacc[nt][r] = v;
        newm[r] = fmaxf(newm[r], v);
      }
    }
#pragma unroll
    for (int off=1; off<16; off<<=1)
#pragma unroll
      for (int r=0;r<8;++r) newm[r] = fmaxf(newm[r], __shfl_xor(newm[r], off, 32));
    float scl[8], ladd[8];
#pragma unroll
    for (int r=0;r<8;++r){ scl[r] = __expf(mrow[r]-newm[r]); mrow[r]=newm[r]; ladd[r]=0.f; }
#pragma unroll
    for (int nt=0;nt<8;++nt)
#pragma unroll
      for (int r=0;r<8;++r){
        float p = __expf(sacc[nt][r] - newm[r]);
        sacc[nt][r] = p;
        ladd[r] += p;
      }
#pragma unroll
    for (int off=1; off<16; off<<=1)
#pragma unroll
      for (int r=0;r<8;++r) ladd[r] += __shfl_xor(ladd[r], off, 32);
#pragma unroll
    for (int r=0;r<8;++r) lrow[r] = lrow[r]*scl[r] + ladd[r];
#pragma unroll
    for (int d=0;d<4;++d)
#pragma unroll
      for (int r=0;r<8;++r) oacc[d][r] *= scl[r];

    // O += P @ V, 32 keys per chunk via per-wave LDS re-staging
#pragma unroll
    for (int ks=0;ks<4;++ks){
#pragma unroll
      for (int t2=0;t2<2;++t2){
        const int nt = ks*2 + t2;
#pragma unroll
        for (int r=0;r<8;++r)
          Ps[wave][half8 + r][t2*16 + l15] = (_Float16)sacc[nt][r];
      }
      asm volatile("s_wait_dscnt 0" ::: "memory");
      v8h plo = *(const v8h*)&Ps[wave][l15][ac0];
      v8h phi = *(const v8h*)&Ps[wave][l15][ac0 + 16];
      v16h pf = mk16(plo, phi);
#pragma unroll
      for (int d=0;d<4;++d){
        const int drow = d*16 + l15;
        const int kcol = ks*32 + bc0;
        v8h vlo = *(const v8h*)&Vt[drow][kcol];
        v8h vhi = *(const v8h*)&Vt[drow][kcol+16];
        v16h vf = mk16(vlo,vhi);
        oacc[d] = __builtin_amdgcn_wmma_f32_16x16x32_f16(false, pf, false, vf, (short)0, oacc[d], false, false);
      }
    }
  }

  // normalize + write attn (B,N,C) f16
#pragma unroll
  for (int d=0; d<4; ++d){
#pragma unroll
    for (int r=0;r<8;++r){
      float o = oacc[d][r] / lrow[r];
      attn16[((size_t)b*N_ + q0 + half8 + r)*C_ + h*D_ + d*16 + l15] = (_Float16)o;
    }
  }
}

// ---------------- LayerNorm over C + transpose to (B,C,H,W) ----------------
__global__ __launch_bounds__(256) void k_ln(const float* __restrict__ resid,
                                            const float* __restrict__ gamma,
                                            const float* __restrict__ beta,
                                            float* __restrict__ out){
  __shared__ float s1[256], s2[256];
  int n = blockIdx.x, b = blockIdx.y, t = threadIdx.x;
  const float* row = resid + ((size_t)b*N_ + n)*C_;
  float v0 = row[t], v1 = row[t+256];
  s1[t] = v0 + v1;
  s2[t] = v0*v0 + v1*v1;
  __syncthreads();
  for (int st=128; st>0; st>>=1){
    if (t<st){ s1[t]+=s1[t+st]; s2[t]+=s2[t+st]; }
    __syncthreads();
  }
  float mu  = s1[0] * (1.0f/(float)C_);
  float var = s2[0] * (1.0f/(float)C_) - mu*mu;
  float rs  = rsqrtf(var + 1e-5f);
  out[((size_t)b*C_ + t)*N_ + n]       = (v0 - mu)*rs*gamma[t]       + beta[t];
  out[((size_t)b*C_ + t + 256)*N_ + n] = (v1 - mu)*rs*gamma[t + 256] + beta[t + 256];
}

extern "C" void kernel_launch(void* const* d_in, const int* in_sizes, int n_in,
                              void* d_out, int out_size, void* d_ws, size_t ws_size,
                              hipStream_t stream) {
  (void)in_sizes; (void)n_in; (void)out_size; (void)ws_size;
  const float* x    = (const float*)d_in[0];
  const float* Wq   = (const float*)d_in[1];
  const float* bq   = (const float*)d_in[2];
  const float* Wk   = (const float*)d_in[3];
  const float* bk   = (const float*)d_in[4];
  const float* Wv   = (const float*)d_in[5];
  const float* bv   = (const float*)d_in[6];
  const float* Wo   = (const float*)d_in[7];
  const float* bo   = (const float*)d_in[8];
  const float* Wsp  = (const float*)d_in[9];
  const float* bsp  = (const float*)d_in[10];
  const float* Wse1 = (const float*)d_in[11];
  const float* bse1 = (const float*)d_in[12];
  const float* Wse2 = (const float*)d_in[13];
  const float* bse2 = (const float*)d_in[14];
  const float* gamma= (const float*)d_in[15];
  const float* beta = (const float*)d_in[16];
  float* out = (float*)d_out;

  // workspace carve-out
  unsigned char* wp = (unsigned char*)d_ws;
  auto carve = [&](size_t bytes) -> void* {
    void* p = (void*)wp;
    wp += (bytes + 255) & ~(size_t)255;
    return p;
  };
  float*    mbuf   = (float*)   carve((size_t)B_*HEADS_*N_*4);
  float*    maskf  = (float*)   carve((size_t)B_*HEADS_*N_*4);
  float*    pooled = (float*)   carve((size_t)B_*C_*4);
  float*    cw     = (float*)   carve((size_t)B_*C_*4);
  float*    xf32   = (float*)   carve((size_t)B_*N_*C_*4);
  _Float16* xf16   = (_Float16*)carve((size_t)B_*N_*C_*2);
  _Float16* wq16   = (_Float16*)carve((size_t)C_*C_*2);
  _Float16* wk16   = (_Float16*)carve((size_t)C_*C_*2);
  _Float16* wv16   = (_Float16*)carve((size_t)C_*C_*2);
  _Float16* wo16   = (_Float16*)carve((size_t)C_*C_*2);
  _Float16* q16    = (_Float16*)carve((size_t)B_*N_*C_*2);
  _Float16* k16    = (_Float16*)carve((size_t)B_*N_*C_*2);
  _Float16* v16b   = (_Float16*)carve((size_t)B_*N_*C_*2);
  _Float16* a16    = (_Float16*)carve((size_t)B_*N_*C_*2);
  float*    resid  = (float*)   carve((size_t)B_*N_*C_*4);

  const int WN = C_*C_;
  k_w2h<<<WN/256, 256, 0, stream>>>(Wq, wq16, WN);
  k_w2h<<<WN/256, 256, 0, stream>>>(Wk, wk16, WN);
  k_w2h<<<WN/256, 256, 0, stream>>>(Wv, wv16, WN);
  k_w2h<<<WN/256, 256, 0, stream>>>(Wo, wo16, WN);

  k_pool<<<B_*C_, 256, 0, stream>>>(x, pooled);
  k_se<<<B_, 256, 0, stream>>>(pooled, Wse1, bse1, Wse2, bse2, cw);
  k_sp<<<(B_*N_)/256, 256, 0, stream>>>(x, Wsp, bsp, mbuf);
  k_mask<<<dim3(HEADS_, B_), 256, 0, stream>>>(mbuf, maskf);
  k_xf<<<dim3(C_/32, N_/32, B_), dim3(32, 8), 0, stream>>>(x, cw, xf32, xf16);

  dim3 ggrid(C_/128, (B_*N_)/128);
  k_gemm<<<ggrid, 256, 0, stream>>>(xf16, wq16, bq, q16, nullptr, nullptr, 0);
  k_gemm<<<ggrid, 256, 0, stream>>>(xf16, wk16, bk, k16, nullptr, nullptr, 0);
  k_gemm<<<ggrid, 256, 0, stream>>>(xf16, wv16, bv, v16b, nullptr, nullptr, 0);

  k_attn<<<dim3(N_/128, HEADS_, B_), 256, 0, stream>>>(q16, k16, v16b, maskf, a16);

  k_gemm<<<ggrid, 256, 0, stream>>>(a16, wo16, bo, nullptr, resid, xf32, 1);

  k_ln<<<dim3(N_, B_), 256, 0, stream>>>(resid, gamma, beta, out);
}
